// EGCL_58755152609326
// MI455X (gfx1250) — matrix operand order repaired and checked by reading, not compiled, not run on hardware
//
#include <hip/hip_runtime.h>
#include <hip/hip_bf16.h>

// ---------------------------------------------------------------------------
// EGCL on MI455X (gfx1250): wave32 WMMA f16->f32 GEMM pipeline.
//   B=16, N=128, NF=H=128.  Layer-1 of the edge MLP is factored into
//   P[b,i]=h_i@eW1a and Q[b,j]=h_j@eW1b+eb1 so the heavy per-edge work is
//   two 128x128x128 GEMMs per (b,i) block, done with v_wmma_f32_16x16x32_f16.
// ---------------------------------------------------------------------------

typedef __attribute__((ext_vector_type(16))) _Float16 v16h;
typedef __attribute__((ext_vector_type(8)))  float    v8f;

#define PITCH   136          // halves per LDS row (K<=128 tiles), even + padded
#define PITCH3  264          // halves per LDS row (K<=256 tiles)
#define TILE_H  (128 * PITCH)
#define TILE_BYTES (TILE_H * 2)

__device__ __forceinline__ float silu_f(float x) {
  return x / (1.0f + __expf(-x));
}

// A-fragment (16x32 f16): lanes 0-15 rows M=0..15 hold K=0-7,16-23;
// lanes 16-31 hold K=8-15,24-31 (two consecutive halves per dword).
__device__ __forceinline__ v16h load_frag_A(const _Float16* base, int row0, int kb, int pitch) {
  const int lane = threadIdx.x & 31;
  const int g = lane >> 4;
  const _Float16* rp = base + (row0 + (lane & 15)) * pitch + kb;
  union { v16h v; unsigned int u[8]; } r;
#pragma unroll
  for (int d = 0; d < 8; ++d) {
    const int ko = ((d >> 2) << 4) + ((d & 3) << 1) + (g << 3);
    r.u[d] = *(const unsigned int*)(rp + ko);
  }
  return r.v;
}

// B-fragment (32x16 f16) gathered from a TRANSPOSED weight tile W^T[N][K]:
// lane n = lane&15; lanes 0-15 hold K=0-7,16-23, lanes 16-31 hold K=8-15,24-31.
__device__ __forceinline__ v16h load_frag_B(const _Float16* baseT, int col0, int kb, int pitch) {
  const int lane = threadIdx.x & 31;
  const int g = lane >> 4;
  const _Float16* rp = baseT + (col0 + (lane & 15)) * pitch + kb;
  union { v16h v; unsigned int u[8]; } r;
#pragma unroll
  for (int d = 0; d < 8; ++d) r.u[d] = *(const unsigned int*)(rp + ((d >> 2) << 4) + ((d & 3) << 1) + (g << 3));
  return r.v;
}

// One wave owns M-tile (wave id) x all 8 N-tiles of a [128 x 128] output.
__device__ __forceinline__ void gemm_tiles(const _Float16* sA, const _Float16* sWt,
                                           int K, int pa, int pw, v8f acc[8]) {
  const int m0 = (threadIdx.x >> 5) << 4;
  for (int kb = 0; kb < K; kb += 32) {
    const v16h a = load_frag_A(sA, m0, kb, pa);
#pragma unroll
    for (int nt = 0; nt < 8; ++nt) {
      const v16h bm = load_frag_B(sWt, nt << 4, kb, pw);
      acc[nt] = __builtin_amdgcn_wmma_f32_16x16x32_f16(false, a, false, bm,
                                                       (short)0, acc[nt], false, false);
    }
  }
}

// D layout: n = lane&15 (+16*nt), m = wave*16 + 8*(lane>>4) + r.
__device__ __forceinline__ void store_act_silu(const v8f acc[8], _Float16* dst, int pitch,
                                               const float* __restrict__ bias) {
  const int lane = threadIdx.x & 31;
  const int g = lane >> 4;
  const int wave = threadIdx.x >> 5;
  const int n_lo = lane & 15;
#pragma unroll
  for (int nt = 0; nt < 8; ++nt) {
    const int n = (nt << 4) + n_lo;
    const float bn = bias[n];
#pragma unroll
    for (int r = 0; r < 8; ++r) {
      const int m = (wave << 4) + (g << 3) + r;
      dst[m * pitch + n] = (_Float16)silu_f(acc[nt][r] + bn);
    }
  }
}

// ---------------------------------------------------------------------------
// Kernel 1: P[b,j,:] = h_j @ eW1[0:128,:],  Q[b,j,:] = h_j @ eW1[128:256,:]+eb1
// grid = B, block = 256 (8 waves).
// ---------------------------------------------------------------------------
__global__ __launch_bounds__(256) void egcl_pq_kernel(
    const float* __restrict__ node_feat, const float* __restrict__ eW1,
    const float* __restrict__ eb1, float* __restrict__ P, float* __restrict__ Q) {
  extern __shared__ __align__(16) unsigned char smem[];
  _Float16* sA = (_Float16*)smem;       // node_feat[b] f16 [128][PITCH]
  _Float16* sW = sA + TILE_H;           // weight^T f16   [128][PITCH]

  const int b = blockIdx.x;
  const int tid = threadIdx.x;
  const int lane = tid & 31, g = lane >> 4, wave = tid >> 5, n_lo = lane & 15;

  for (int idx = tid; idx < 128 * 128; idx += 256) {
    const int j = idx >> 7, c = idx & 127;
    sA[j * PITCH + c] = (_Float16)node_feat[(b * 128 + j) * 128 + c];
  }
  for (int idx = tid; idx < 128 * 128; idx += 256) {
    const int k = idx >> 7, n = idx & 127;
    sW[n * PITCH + k] = (_Float16)eW1[k * 128 + n];      // rows 0..127
  }
  __syncthreads();
  {
    v8f acc[8] = {};
    gemm_tiles(sA, sW, 128, PITCH, PITCH, acc);
#pragma unroll
    for (int nt = 0; nt < 8; ++nt) {
      const int n = (nt << 4) + n_lo;
#pragma unroll
      for (int r = 0; r < 8; ++r) {
        const int m = (wave << 4) + (g << 3) + r;
        P[(b * 128 + m) * 128 + n] = acc[nt][r];
      }
    }
  }
  __syncthreads();
  for (int idx = tid; idx < 128 * 128; idx += 256) {
    const int k = idx >> 7, n = idx & 127;
    sW[n * PITCH + k] = (_Float16)eW1[(128 + k) * 128 + n]; // rows 128..255
  }
  __syncthreads();
  {
    v8f acc[8] = {};
    gemm_tiles(sA, sW, 128, PITCH, PITCH, acc);
#pragma unroll
    for (int nt = 0; nt < 8; ++nt) {
      const int n = (nt << 4) + n_lo;
      const float bn = eb1[n];
#pragma unroll
      for (int r = 0; r < 8; ++r) {
        const int m = (wave << 4) + (g << 3) + r;
        Q[(b * 128 + m) * 128 + n] = acc[nt][r] + bn;
      }
    }
  }
}

// ---------------------------------------------------------------------------
// Kernel 2: one block per (b,i).  Builds t1[j,h] = P_i + Q_j + dist*wd + adj*wa,
// SiLU; GEMM @eW2 + SiLU -> m; colsum -> message; GEMM @cW1 + SiLU; GEMV cW2
// -> phi_x; coordinate reduction -> pos_out.
// ---------------------------------------------------------------------------
__global__ __launch_bounds__(256) void egcl_edge_kernel(
    const float* __restrict__ pos, const float* __restrict__ valid,
    const float* __restrict__ adj, const float* __restrict__ eW1,
    const float* __restrict__ eW2, const float* __restrict__ eb2,
    const float* __restrict__ cW1, const float* __restrict__ cb1,
    const float* __restrict__ cW2, const float* __restrict__ cb2,
    const float* __restrict__ P, const float* __restrict__ Q,
    float* __restrict__ msg_out, float* __restrict__ pos_out) {
  extern __shared__ __align__(16) unsigned char smem[];
  _Float16* sA0  = (_Float16*)smem;       // activations ping
  _Float16* sA1  = sA0 + TILE_H;          // activations pong
  _Float16* sW   = sA1 + TILE_H;          // current weight^T
  float* sDist = (float*)(sW + TILE_H);   // 128
  float* sPhi  = sDist + 128;             // 128
  float* sPi   = sPhi + 128;              // 128
  float* sWd   = sPi + 128;               // 128
  float* sWa   = sWd + 128;               // 128
  float* sPos  = sWa + 128;               // 4

  const int tid = threadIdx.x;
  const int b = blockIdx.x >> 7;
  const int i = blockIdx.x & 127;
  const int bi = b * 128 + i;

  if (tid < 128) {
    sPi[tid] = P[bi * 128 + tid];
    sWd[tid] = eW1[256 * 128 + tid];
    sWa[tid] = eW1[257 * 128 + tid];
  }
  if (tid < 3) sPos[tid] = pos[bi * 3 + tid];
  __syncthreads();

  if (tid < 128) {
    const int j = tid;
    const float dx = sPos[0] - pos[(b * 128 + j) * 3 + 0];
    const float dy = sPos[1] - pos[(b * 128 + j) * 3 + 1];
    const float dz = sPos[2] - pos[(b * 128 + j) * 3 + 2];
    sDist[j] = dx * dx + dy * dy + dz * dz;
  }
  for (int idx = tid; idx < 128 * 128; idx += 256) {   // eW2^T into LDS
    const int k = idx >> 7, n = idx & 127;
    sW[n * PITCH + k] = (_Float16)eW2[idx];
  }
  __syncthreads();

  const float vmask = valid[bi];
  for (int idx = tid; idx < 128 * 128; idx += 256) {   // layer-1 activation
    const int j = idx >> 7, h = idx & 127;
    const float t = sPi[h] + Q[(b * 128 + j) * 128 + h] +
                    sDist[j] * sWd[h] + adj[bi * 128 + j] * sWa[h];
    sA0[j * PITCH + h] = (_Float16)silu_f(t);
  }
  __syncthreads();

  {                                                    // m = SiLU(h1 @ eW2 + eb2)
    v8f acc[8] = {};
    gemm_tiles(sA0, sW, 128, PITCH, PITCH, acc);
    store_act_silu(acc, sA1, PITCH, eb2);
  }
  __syncthreads();

  if (tid < 128) {                                     // message = valid * colsum(m)
    float s = 0.0f;
    for (int j = 0; j < 128; ++j) s += (float)sA1[j * PITCH + tid];
    msg_out[bi * 128 + tid] = vmask * s;
  }
  for (int idx = tid; idx < 128 * 128; idx += 256) {   // cW1^T into LDS
    const int k = idx >> 7, n = idx & 127;
    sW[n * PITCH + k] = (_Float16)cW1[idx];
  }
  __syncthreads();

  {                                                    // mid = SiLU(m @ cW1 + cb1)
    v8f acc[8] = {};
    gemm_tiles(sA1, sW, 128, PITCH, PITCH, acc);
    store_act_silu(acc, sA0, PITCH, cb1);
  }
  __syncthreads();

  if (tid < 128) {                                     // phi_x = mid @ cW2 + cb2
    const int j = tid;
    float dot = cb2[0];
    for (int h = 0; h < 128; ++h) dot += (float)sA0[j * PITCH + h] * cW2[h];
    sPhi[j] = dot;
  }
  __syncthreads();

  if (tid < 3) {                                       // pos_out
    const int d = tid;
    float s = 0.0f;
    for (int j = 0; j < 128; ++j)
      s += (sPos[d] - pos[(b * 128 + j) * 3 + d]) * sPhi[j];
    pos_out[bi * 3 + d] = sPos[d] + vmask * s;
  }
}

// ---------------------------------------------------------------------------
// Kernel 3: node MLP.  node_out = SiLU([h|msg] @ nW1 + nb1) @ nW2 + nb2.
// grid = B, block = 256.
// ---------------------------------------------------------------------------
__global__ __launch_bounds__(256) void egcl_node_kernel(
    const float* __restrict__ node_feat, const float* __restrict__ msg,
    const float* __restrict__ nW1, const float* __restrict__ nb1,
    const float* __restrict__ nW2, const float* __restrict__ nb2,
    float* __restrict__ node_out) {
  extern __shared__ __align__(16) unsigned char smem[];
  _Float16* sIn = (_Float16*)smem;        // [128][PITCH3] input (reused as mid)
  _Float16* sWt = sIn + 128 * PITCH3;     // [128][PITCH3] weight^T

  const int b = blockIdx.x;
  const int tid = threadIdx.x;
  const int lane = tid & 31, g = lane >> 4, wave = tid >> 5, n_lo = lane & 15;

  for (int idx = tid; idx < 128 * 256; idx += 256) {
    const int j = idx >> 8, c = idx & 255;
    const float v = (c < 128) ? node_feat[(b * 128 + j) * 128 + c]
                              : msg[(b * 128 + j) * 128 + (c - 128)];
    sIn[j * PITCH3 + c] = (_Float16)v;
  }
  for (int idx = tid; idx < 256 * 128; idx += 256) {
    const int k = idx >> 7, n = idx & 127;
    sWt[n * PITCH3 + k] = (_Float16)nW1[idx];
  }
  __syncthreads();

  v8f acc[8] = {};
  gemm_tiles(sIn, sWt, 256, PITCH3, PITCH3, acc);
  __syncthreads();                        // all waves done reading sIn
  store_act_silu(acc, sIn, PITCH, nb1);   // mid reuses sIn region, pitch 136
  __syncthreads();

  for (int idx = tid; idx < 128 * 128; idx += 256) {
    const int k = idx >> 7, n = idx & 127;
    sWt[n * PITCH3 + k] = (_Float16)nW2[idx];
  }
  __syncthreads();

  v8f acc2[8] = {};
  gemm_tiles(sIn, sWt, 128, PITCH, PITCH3, acc2);
#pragma unroll
  for (int nt = 0; nt < 8; ++nt) {
    const int n = (nt << 4) + n_lo;
    const float bn = nb2[n];
#pragma unroll
    for (int r = 0; r < 8; ++r) {
      const int m = (wave << 4) + (g << 3) + r;
      node_out[(b * 128 + m) * 128 + n] = acc2[nt][r] + bn;
    }
  }
}

// ---------------------------------------------------------------------------
extern "C" void kernel_launch(void* const* d_in, const int* in_sizes, int n_in,
                              void* d_out, int out_size, void* d_ws, size_t ws_size,
                              hipStream_t stream) {
  (void)in_sizes; (void)n_in; (void)out_size; (void)ws_size;
  const float* node_feat = (const float*)d_in[0];
  const float* pos       = (const float*)d_in[1];
  const float* valid     = (const float*)d_in[2];
  const float* adj       = (const float*)d_in[3];
  const float* eW1 = (const float*)d_in[4];
  const float* eb1 = (const float*)d_in[5];
  const float* eW2 = (const float*)d_in[6];
  const float* eb2 = (const float*)d_in[7];
  const float* cW1 = (const float*)d_in[8];
  const float* cb1 = (const float*)d_in[9];
  const float* cW2 = (const float*)d_in[10];
  const float* cb2 = (const float*)d_in[11];
  const float* nW1 = (const float*)d_in[12];
  const float* nb1 = (const float*)d_in[13];
  const float* nW2 = (const float*)d_in[14];
  const float* nb2 = (const float*)d_in[15];

  float* out      = (float*)d_out;
  float* pos_out  = out;                       // [16,128,3]  = 6144 floats
  float* node_out = out + 16 * 128 * 3;        // [16,128,128]

  float* wsP = (float*)d_ws;                   // [16,128,128]
  float* wsQ = wsP + 16 * 128 * 128;           // [16,128,128]
  float* wsM = wsQ + 16 * 128 * 128;           // [16,128,128] message

  const size_t smem1 = 2 * (size_t)TILE_BYTES;                         // ~70 KB
  const size_t smem2 = 3 * (size_t)TILE_BYTES + (5 * 128 + 4) * 4;     // ~107 KB
  const size_t smem3 = 2 * (size_t)(128 * PITCH3 * 2);                 // ~135 KB

  egcl_pq_kernel<<<16, 256, smem1, stream>>>(node_feat, eW1, eb1, wsP, wsQ);
  egcl_edge_kernel<<<16 * 128, 256, smem2, stream>>>(
      pos, valid, adj, eW1, eW2, eb2, cW1, cb1, cW2, cb2, wsP, wsQ, wsM, pos_out);
  egcl_node_kernel<<<16, 256, smem3, stream>>>(node_feat, wsM, nW1, nb1, nW2, nb2,
                                               node_out);
}
